// DistanceDiscriminator_21766894256326
// MI455X (gfx1250) — matrix operand order repaired and checked by reading, not compiled, not run on hardware
//
#include <hip/hip_runtime.h>
#include <hip/hip_bf16.h>

// ---------------------------------------------------------------------------
// DistanceDiscriminator fused kernel for MI455X (gfx1250, wave32).
//
// out[N,64] = log(sqrt(max(Q[d] - 2*x*S[d] + N*x^2, 0)) + eps) @ W^T + b
//
// Memory-bound problem (~128 MB HBM traffic, ~5.5 us floor at 23.3 TB/s);
// GEMM part is only ~2.1 GFLOP, so WMMA precision is chosen for accuracy:
// bf16 hi/lo decomposition (3x v_wmma_f32_16x16x32_bf16 per 16x16x32 tile)
// gives near-fp32 dot products at negligible extra cost.
// ---------------------------------------------------------------------------

#define N_ROWS 4096
#define D_DIM  4096
#define OUT_N  64
#define KC     256          // staged K-chunk per workgroup
#define EPSV   1e-6f

typedef __attribute__((ext_vector_type(16))) __bf16        v16bf;
typedef __attribute__((ext_vector_type(8)))  float         v8f;
typedef __attribute__((ext_vector_type(4)))  unsigned int  v4u;

union ABfrag { v16bf bf; v4u u[2]; };

// ---------------------------------------------------------------------------
// Stage 1: partial column sums S/Q over 256-row strips (coalesced).
// gridDim = (D/256, 16), block = 256.
// ---------------------------------------------------------------------------
__global__ __launch_bounds__(256) void k_colsum(const float* __restrict__ x,
                                                float* __restrict__ partS,
                                                float* __restrict__ partQ) {
    const int col   = blockIdx.x * 256 + threadIdx.x;
    const int chunk = blockIdx.y;                    // 0..15
    const size_t r0 = (size_t)chunk * 256;
    const float* p  = x + r0 * D_DIM + col;
    float s = 0.0f, q = 0.0f;
    #pragma unroll 8
    for (int r = 0; r < 256; ++r) {
        float v = p[(size_t)r * D_DIM];
        s += v;
        q = fmaf(v, v, q);
    }
    partS[chunk * D_DIM + col] = s;
    partQ[chunk * D_DIM + col] = q;
}

// ---------------------------------------------------------------------------
// Stage 2: deterministic fixed-order reduction of the 16 partials per column.
// ---------------------------------------------------------------------------
__global__ __launch_bounds__(256) void k_reduce(const float* __restrict__ partS,
                                                const float* __restrict__ partQ,
                                                float* __restrict__ S,
                                                float* __restrict__ Q) {
    const int col = blockIdx.x * 256 + threadIdx.x;
    float s = 0.0f, q = 0.0f;
    #pragma unroll
    for (int c = 0; c < 16; ++c) {
        s += partS[c * D_DIM + col];
        q += partQ[c * D_DIM + col];
    }
    S[col] = s;
    Q[col] = q;
}

// ---------------------------------------------------------------------------
// Stage 3: split W (f32, row-major [64][4096]) into bf16 hi/lo planes.
// ---------------------------------------------------------------------------
__global__ __launch_bounds__(256) void k_wsplit(const float* __restrict__ W,
                                                __bf16* __restrict__ Whi,
                                                __bf16* __restrict__ Wlo) {
    const int i = blockIdx.x * 256 + threadIdx.x;    // < 64*4096
    const float w  = W[i];
    const __bf16 hi = (__bf16)w;
    const float  lo = w - (float)hi;
    Whi[i] = hi;
    Wlo[i] = (__bf16)lo;
}

// ---------------------------------------------------------------------------
// Stage 4: fused map + GEMM. One workgroup (256 thr = 8 wave32) per 16-row
// output tile; K=4096 split across the 8 waves. Per KC=256 chunk:
//   Phase A: coalesced fp32 loads of x[16][256], fused sqrt/log, bf16 hi/lo
//            planes staged in padded LDS.
//   Phase B: wave w takes K-subchunk w*32; 4 N-tiles x 3 WMMAs (bf16x3).
// Cross-wave reduction is wave-ordered in LDS (deterministic, no atomics).
// ---------------------------------------------------------------------------
__global__ __launch_bounds__(256) void k_main(const float* __restrict__ x,
                                              const float* __restrict__ S,
                                              const float* __restrict__ Q,
                                              const __bf16* __restrict__ Whi,
                                              const __bf16* __restrict__ Wlo,
                                              const float* __restrict__ bias,
                                              float* __restrict__ out) {
    // +8 bf16 row padding => row stride 132 dwords -> rotates LDS banks.
    __shared__ __align__(16) __bf16 sAhi[16][KC + 8];
    __shared__ __align__(16) __bf16 sAlo[16][KC + 8];
    __shared__ float sC[16][OUT_N];

    const int tid  = threadIdx.x;
    const int lane = tid & 31;
    const int wave = tid >> 5;
    const int row0 = blockIdx.x * 16;

    // zero the cross-wave accumulator tile
    for (int i = tid; i < 16 * OUT_N; i += 256)
        ((float*)sC)[i] = 0.0f;

    // CDNA5 16-bit A-fragment addressing (16x32):
    //   lanes 0-15 : row=lane,    K = {0..7, 16..23}
    //   lanes 16-31: row=lane-16, K = {8..15, 24..31}
    const int m    = lane & 15;
    const int aoff = (lane < 16) ? 0 : 8;
    const int kloc = wave * 32;                 // this wave's K-subchunk in LDS
    // B-fragment (32x16): lanes 0-15 hold K {0..7, 16..23}, lanes 16-31 hold K {8..15, 24..31}
    const int khalf = (lane >> 4) << 3;

    v8f acc[4] = {v8f{}, v8f{}, v8f{}, v8f{}};

    for (int kb = 0; kb < D_DIM; kb += KC) {
        // ---- Phase A: load + fused transform into LDS bf16 planes ----
        const float Sv = S[kb + tid];
        const float Qv = Q[kb + tid];
        #pragma unroll
        for (int e = 0; e < 16; ++e) {
            const float xv = x[(size_t)(row0 + e) * D_DIM + kb + tid];
            // Q - 2*x*S + N*x^2  ==  fma(x, fma(N, x, -2S), Q)
            const float t  = fmaf((float)N_ROWS, xv, -2.0f * Sv);
            const float sq = fmaf(xv, t, Qv);
            const float lg = logf(sqrtf(fmaxf(sq, 0.0f)) + EPSV);
            const __bf16 hi = (__bf16)lg;
            sAhi[e][tid] = hi;
            sAlo[e][tid] = (__bf16)(lg - (float)hi);
        }
        if (kb + KC < D_DIM)    // hint next chunk into cache (global_prefetch_b8)
            __builtin_prefetch(&x[(size_t)row0 * D_DIM + kb + KC + tid], 0, 0);
        __syncthreads();

        // ---- Phase B: WMMA on this wave's 16x32 sub-chunk ----
        ABfrag ahi, alo;
        ahi.u[0] = *(const v4u*)&sAhi[m][kloc + aoff];
        ahi.u[1] = *(const v4u*)&sAhi[m][kloc + aoff + 16];
        alo.u[0] = *(const v4u*)&sAlo[m][kloc + aoff];
        alo.u[1] = *(const v4u*)&sAlo[m][kloc + aoff + 16];

        const int kg = kb + kloc + khalf;       // global K index for B frag
        #pragma unroll
        for (int nt = 0; nt < 4; ++nt) {
            const int n = nt * 16 + m;          // output column for this lane
            const __bf16* ph = Whi + (size_t)n * D_DIM + kg;
            const __bf16* pl = Wlo + (size_t)n * D_DIM + kg;
            ABfrag bhi, blo;
            bhi.u[0] = *(const v4u*)ph;  bhi.u[1] = *(const v4u*)(ph + 16);
            blo.u[0] = *(const v4u*)pl;  blo.u[1] = *(const v4u*)(pl + 16);

            // bf16x3: hi*hi + hi*lo + lo*hi  (lo*lo below fp32 ulp)
            acc[nt] = __builtin_amdgcn_wmma_f32_16x16x32_bf16(
                false, ahi.bf, false, bhi.bf, (short)0, acc[nt], false, false);
            acc[nt] = __builtin_amdgcn_wmma_f32_16x16x32_bf16(
                false, ahi.bf, false, blo.bf, (short)0, acc[nt], false, false);
            acc[nt] = __builtin_amdgcn_wmma_f32_16x16x32_bf16(
                false, alo.bf, false, bhi.bf, (short)0, acc[nt], false, false);
        }
        __syncthreads();                        // LDS A reused next chunk
    }

    // ---- deterministic wave-ordered K reduction into sC ----
    // C layout (16x16 f32): VGPR v -> row v (lanes 0-15) / v+8 (lanes 16-31),
    // N = lane & 15.
    const int rbase = (lane >> 4) << 3;
    for (int w8 = 0; w8 < 8; ++w8) {
        if (wave == w8) {
            #pragma unroll
            for (int nt = 0; nt < 4; ++nt) {
                #pragma unroll
                for (int v = 0; v < 8; ++v)
                    sC[rbase + v][nt * 16 + m] += acc[nt][v];
            }
        }
        __syncthreads();
    }

    // ---- bias + store ----
    #pragma unroll
    for (int j = 0; j < 4; ++j) {
        const int idx = tid * 4 + j;
        const int r = idx >> 6, c = idx & 63;
        out[(size_t)(row0 + r) * OUT_N + c] = sC[r][c] + bias[c];
    }
}

// ---------------------------------------------------------------------------
// Workspace layout (bytes):
//   S      : 4096 f32                  @ 0
//   Q      : 4096 f32                  @ 16 KB
//   partS  : 16*4096 f32               @ 32 KB
//   partQ  : 16*4096 f32               @ 288 KB
//   Whi    : 64*4096 bf16              @ 544 KB
//   Wlo    : 64*4096 bf16              @ 1056 KB
// Total ~1.57 MB.
// ---------------------------------------------------------------------------
extern "C" void kernel_launch(void* const* d_in, const int* in_sizes, int n_in,
                              void* d_out, int out_size, void* d_ws, size_t ws_size,
                              hipStream_t stream) {
    const float* x = (const float*)d_in[0];
    const float* W = (const float*)d_in[1];
    const float* b = (const float*)d_in[2];
    float* out = (float*)d_out;

    float*  S     = (float*)d_ws;
    float*  Q     = S + D_DIM;
    float*  partS = Q + D_DIM;
    float*  partQ = partS + 16 * D_DIM;
    __bf16* Whi   = (__bf16*)(partQ + 16 * D_DIM);
    __bf16* Wlo   = Whi + (size_t)OUT_N * D_DIM;

    k_colsum<<<dim3(D_DIM / 256, 16), 256, 0, stream>>>(x, partS, partQ);
    k_reduce<<<D_DIM / 256, 256, 0, stream>>>(partS, partQ, S, Q);
    k_wsplit<<<(OUT_N * D_DIM) / 256, 256, 0, stream>>>(W, Whi, Wlo);
    k_main<<<N_ROWS / 16, 256, 0, stream>>>(x, S, Q, Whi, Wlo, b, out);
}
